// Model_25451976196676
// MI455X (gfx1250) — matrix operand order repaired and checked by reading, not compiled, not run on hardware
//
#include <hip/hip_runtime.h>
#include <math.h>

#define SEQ  2048
#define BSZ  4
#define DIM  512
#define NH   8
#define NR   8
#define QKD  32
#define FFND 2048
#define HDM  64
#define MROWS (BSZ * SEQ)

typedef __attribute__((ext_vector_type(16))) __bf16 v16bf;
typedef __attribute__((ext_vector_type(8)))  float  v8f;
typedef __attribute__((ext_vector_type(4)))  int    vi4;

// ---- CDNA5 async global->LDS path (probe-guarded, safe fallback) ----
#ifndef USE_ASYNC_LDS
#define USE_ASYNC_LDS 1
#endif
#if USE_ASYNC_LDS && defined(__has_builtin)
#if __has_builtin(__builtin_amdgcn_global_load_async_to_lds_b128)
#define HAVE_ASYNC 1
#endif
#endif
#ifndef HAVE_ASYNC
#define HAVE_ASYNC 0
#endif
// builtin signature (from compiler diagnostic): (v4i addrspace(1)*, v4i addrspace(3)*, imm, imm)
#define AS1V4(p) ((__attribute__((address_space(1))) vi4*)(p))
#define AS3V4(p) ((__attribute__((address_space(3))) vi4*)(p))

__device__ __forceinline__ void async_wait_all() {
#if HAVE_ASYNC
#if defined(__has_builtin) && __has_builtin(__builtin_amdgcn_s_wait_asynccnt)
  __builtin_amdgcn_s_wait_asynccnt(0);
#else
  asm volatile("s_wait_asynccnt 0" ::: "memory");
#endif
#endif
}

__device__ __forceinline__ unsigned short f2bf(float f) {
  unsigned int u = __float_as_uint(f);
  u += 0x7FFFu + ((u >> 16) & 1u);
  return (unsigned short)(u >> 16);
}
__device__ __forceinline__ float bf2f(unsigned short h) {
  return __uint_as_float(((unsigned int)h) << 16);
}

// Fragment-major LDS: each fragment = 32 lanes * 16 bf16 (32B contiguous per lane).
__device__ __forceinline__ v16bf lds_frag(const unsigned short* base, int frag, int lane) {
  return *(const v16bf*)(base + frag * 512 + lane * 16);
}
__device__ __forceinline__ v8f wmma_bf16(v16bf a, v16bf b, v8f c) {
  return __builtin_amdgcn_wmma_f32_16x16x32_bf16(false, a, false, b, (short)0, c, false, false);
}
// A-operand (16x32 bf16) lane/slot for element (m, kk)  (used for P scatter)
__device__ __forceinline__ int a_lane(int m, int kk) { return m + 16 * (kk >> 4); }
__device__ __forceinline__ int a_slot(int kk) { return kk & 15; }

// ---------------------------------------------------------------- pack kernels
__global__ void k_pack_x(const float* __restrict__ x, unsigned short* __restrict__ xb) {
  int idx = blockIdx.x * blockDim.x + threadIdx.x;   // over MROWS*DIM
  int c = idx & (DIM - 1);
  int row = idx / DIM;                                // row = b*SEQ + s
  int b = row / SEQ, s = row % SEQ;
  xb[idx] = f2bf(x[((size_t)s * BSZ + b) * DIM + c]); // x is [S,B,D]
}
// transpose-pack: o[n*K + k] = bf16(w[k*N + n])   (weights become [N,K] row-major)
__global__ void k_pack_wT(const float* __restrict__ w, unsigned short* __restrict__ o,
                          int K, int N) {
  int idx = blockIdx.x * blockDim.x + threadIdx.x;
  if (idx < K * N) {
    int n = idx / K, k = idx - n * K;
    o[idx] = f2bf(w[(size_t)k * N + n]);
  }
}

// ---------------------------------------------------------------- generic GEMM
// C[M,N] = epilogue( A[M,K](bf16, row-major) * Bt[N,K](bf16, pre-transposed) + bias )
// 64x64 tile, 4 waves. mode 0: bf16 ; 1: relu+bf16 ; 2: f32 ; 3: bf16 transposed (col*MROWS+row)
__device__ __forceinline__ void gemm_store(const v8f& ac, int rowTop, int col, int Ndim,
                                           float bvv, float scale, int mode,
                                           unsigned short* Cb, float* Cf, int hL) {
#pragma unroll
  for (int r = 0; r < 8; ++r) {
    int row = rowTop + r + 8 * hL;
    float v = (ac[r] + bvv) * scale;
    if (mode == 1) v = fmaxf(v, 0.f);
    if (mode == 2)      Cf[(size_t)row * Ndim + col] = v;
    else if (mode == 3) Cb[(size_t)col * MROWS + row] = f2bf(v);
    else                Cb[(size_t)row * Ndim + col] = f2bf(v);
  }
}

__global__ __launch_bounds__(128)
void k_gemm(const unsigned short* __restrict__ A, const unsigned short* __restrict__ Bt,
            const float* __restrict__ bias, float scale, int Kdim, int Ndim, int mode,
            unsigned short* __restrict__ Cb, float* __restrict__ Cf) {
  __shared__ __align__(32) unsigned short sA[4 * 512];
  __shared__ __align__(32) unsigned short sB[4 * 512];
  const int tid = threadIdx.x;
  const int wave = tid >> 5, lane = tid & 31;
  const int wm = wave >> 1, wn = wave & 1;
  const int rowBase = blockIdx.y * 64;
  const int colBase = blockIdx.x * 64;
  v8f acc00 = {}, acc01 = {}, acc10 = {}, acc11 = {};

  for (int k0 = 0; k0 < Kdim; k0 += 32) {
    __syncthreads();
    // A tile 64x32 -> A-frag layout: 256 x 16B vector copies
#pragma unroll
    for (int e = tid; e < 256; e += 128) {
      int m = e >> 2, h = (e >> 1) & 1, part = e & 1;
      const uint4 v = *(const uint4*)(A + (size_t)(rowBase + m) * Kdim + k0 + h * 16 + part * 8);
      *(uint4*)(sA + (m >> 4) * 512 + ((m & 15) + 16 * h) * 16 + part * 8) = v;
    }
    // B tile (from Bt[N,K]) -> B-frag layout: 128 x 32B vector copies
    {
      int n = tid >> 1, h2 = tid & 1;
      const uint4* g = (const uint4*)(Bt + (size_t)(colBase + n) * Kdim + k0 + h2 * 16);
      uint4* d = (uint4*)(sB + (n >> 4) * 512 + ((n & 15) + 16 * h2) * 16);
      d[0] = g[0]; d[1] = g[1];
    }
    if (k0 + 32 < Kdim) {                        // CDNA5 global_prefetch_b8 of next tiles
      __builtin_prefetch(A + (size_t)(rowBase + (tid >> 1)) * Kdim + k0 + 32, 0, 3);
      __builtin_prefetch(Bt + (size_t)(colBase + (tid >> 1)) * Kdim + k0 + 32, 0, 3);
    }
    __syncthreads();
    v16bf a0 = lds_frag(sA, 2 * wm,     lane);
    v16bf a1 = lds_frag(sA, 2 * wm + 1, lane);
    v16bf b0 = lds_frag(sB, 2 * wn,     lane);
    v16bf b1 = lds_frag(sB, 2 * wn + 1, lane);
    acc00 = wmma_bf16(a0, b0, acc00);
    acc01 = wmma_bf16(a0, b1, acc01);
    acc10 = wmma_bf16(a1, b0, acc10);
    acc11 = wmma_bf16(a1, b1, acc11);
  }

  const int hL = lane >> 4, nn = lane & 15;
  int c0 = colBase + (2 * wn) * 16 + nn;
  int c1 = colBase + (2 * wn + 1) * 16 + nn;
  float bv0 = bias ? bias[c0] : 0.f;
  float bv1 = bias ? bias[c1] : 0.f;
  gemm_store(acc00, rowBase + (2 * wm) * 16,     c0, Ndim, bv0, scale, mode, Cb, Cf, hL);
  gemm_store(acc01, rowBase + (2 * wm) * 16,     c1, Ndim, bv1, scale, mode, Cb, Cf, hL);
  gemm_store(acc10, rowBase + (2 * wm + 1) * 16, c0, Ndim, bv0, scale, mode, Cb, Cf, hL);
  gemm_store(acc11, rowBase + (2 * wm + 1) * 16, c1, Ndim, bv1, scale, mode, Cb, Cf, hL);
}

// ------------------------------------------------- attention pass A: softmax stats
__global__ __launch_bounds__(128)
void k_attn_stats(const unsigned short* __restrict__ Q16, const unsigned short* __restrict__ K16,
                  float* __restrict__ mbuf, float* __restrict__ lbuf) {
  __shared__ __align__(32) unsigned short sQ[8 * 512];
  __shared__ __align__(32) unsigned short sK[8 * 512];
  const int bh = blockIdx.y, b = bh / NH, hq = bh % NH;
  const int row0 = blockIdx.x * 64;
  const int tid = threadIdx.x, wave = tid >> 5, lane = tid & 31;

  // Q tile 64x64 -> A-frags: 512 x 16B copies
#pragma unroll
  for (int e = tid; e < 512; e += 128) {
    int m = e >> 3, c = (e >> 2) & 1, h = (e >> 1) & 1, part = e & 1;
    int dbase = c * 32 + h * 16 + part * 8;
    const uint4 v = *(const uint4*)(Q16 + (size_t)(b * SEQ + row0 + m) * DIM + hq * HDM + dbase);
    *(uint4*)(sQ + ((m >> 4) * 2 + c) * 512 + ((m & 15) + 16 * h) * 16 + part * 8) = v;
  }
  __syncthreads();
  v16bf q0 = lds_frag(sQ, wave * 2,     lane);
  v16bf q1 = lds_frag(sQ, wave * 2 + 1, lane);

  float mrun[8], lrun[8];
#pragma unroll
  for (int r = 0; r < 8; ++r) { mrun[r] = -1e30f; lrun[r] = 0.f; }

  for (int t0 = 0; t0 < SEQ; t0 += 64) {
    __syncthreads();
    // K tile -> B-frags (K rows already contiguous in d): 256 x 32B copies
#pragma unroll
    for (int e = tid; e < 256; e += 128) {
      int key = e >> 2, c = (e >> 1) & 1, h2 = e & 1;
      int dbase = c * 32 + h2 * 16;
      const uint4* g = (const uint4*)(K16 + (size_t)(b * SEQ + t0 + key) * DIM + hq * HDM + dbase);
      uint4* d = (uint4*)(sK + ((key >> 4) * 2 + c) * 512 + ((key & 15) + 16 * h2) * 16);
      d[0] = g[0]; d[1] = g[1];
    }
    if (t0 + 64 < SEQ)
      __builtin_prefetch(K16 + (size_t)(b * SEQ + t0 + 64 + (tid >> 1)) * DIM + hq * HDM, 0, 3);
    __syncthreads();
    v8f sc0 = {}, sc1 = {}, sc2 = {}, sc3 = {};
    sc0 = wmma_bf16(q0, lds_frag(sK, 0, lane), sc0);
    sc0 = wmma_bf16(q1, lds_frag(sK, 1, lane), sc0);
    sc1 = wmma_bf16(q0, lds_frag(sK, 2, lane), sc1);
    sc1 = wmma_bf16(q1, lds_frag(sK, 3, lane), sc1);
    sc2 = wmma_bf16(q0, lds_frag(sK, 4, lane), sc2);
    sc2 = wmma_bf16(q1, lds_frag(sK, 5, lane), sc2);
    sc3 = wmma_bf16(q0, lds_frag(sK, 6, lane), sc3);
    sc3 = wmma_bf16(q1, lds_frag(sK, 7, lane), sc3);
#pragma unroll
    for (int r = 0; r < 8; ++r) {
      float mx = fmaxf(fmaxf(sc0[r], sc1[r]), fmaxf(sc2[r], sc3[r]));
      mx = fmaxf(mx, __shfl_xor(mx, 1, 32));
      mx = fmaxf(mx, __shfl_xor(mx, 2, 32));
      mx = fmaxf(mx, __shfl_xor(mx, 4, 32));
      mx = fmaxf(mx, __shfl_xor(mx, 8, 32));
      float nm = fmaxf(mrun[r], mx);
      float se = __expf(sc0[r] - nm) + __expf(sc1[r] - nm) +
                 __expf(sc2[r] - nm) + __expf(sc3[r] - nm);
      se += __shfl_xor(se, 1, 32);
      se += __shfl_xor(se, 2, 32);
      se += __shfl_xor(se, 4, 32);
      se += __shfl_xor(se, 8, 32);
      lrun[r] = lrun[r] * __expf(mrun[r] - nm) + se;
      mrun[r] = nm;
    }
  }
  if ((lane & 15) == 0) {
    int hL = lane >> 4;
#pragma unroll
    for (int r = 0; r < 8; ++r) {
      int row = row0 + wave * 16 + r + 8 * hL;
      mbuf[(size_t)bh * SEQ + row] = mrun[r];
      lbuf[(size_t)bh * SEQ + row] = lrun[r];
    }
  }
}

// ------------------------------------------------- attention pass B: out = P @ V
// V supplied pre-transposed: Vt[col][row] with row = b*SEQ+s (ld = MROWS)
__global__ __launch_bounds__(256)
void k_attn_out(const unsigned short* __restrict__ Q16, const unsigned short* __restrict__ K16,
                const unsigned short* __restrict__ Vt, const float* __restrict__ mbuf,
                const float* __restrict__ lbuf, unsigned short* __restrict__ out16) {
  __shared__ __align__(32) unsigned short sQ[8 * 512];
  __shared__ __align__(32) unsigned short sK[8 * 512];
  __shared__ __align__(32) unsigned short sP[8 * 512];
  __shared__ __align__(32) unsigned short sV[32 * 512];   // 64 keys x 256 cols, B-frags
  const int bh = blockIdx.y, b = bh / NH, hq = bh % NH;
  const int row0 = blockIdx.x * 64;
  const int colbase = blockIdx.z * 256;
  const int tid = threadIdx.x, wave = tid >> 5, lane = tid & 31;
  const int wr = wave >> 1, wc = wave & 1;
  const int hL = lane >> 4, nn = lane & 15;

#pragma unroll
  for (int e = tid; e < 512; e += 256) {        // Q -> A frags (16B copies)
    int m = e >> 3, c = (e >> 2) & 1, h = (e >> 1) & 1, part = e & 1;
    int dbase = c * 32 + h * 16 + part * 8;
    const uint4 v = *(const uint4*)(Q16 + (size_t)(b * SEQ + row0 + m) * DIM + hq * HDM + dbase);
    *(uint4*)(sQ + ((m >> 4) * 2 + c) * 512 + ((m & 15) + 16 * h) * 16 + part * 8) = v;
  }
  float mi[8], li[8];
#pragma unroll
  for (int r = 0; r < 8; ++r) {
    int row = row0 + wr * 16 + r + 8 * hL;
    mi[r] = mbuf[(size_t)bh * SEQ + row];
    li[r] = 1.f / lbuf[(size_t)bh * SEQ + row];
  }
  __syncthreads();
  v16bf q0 = lds_frag(sQ, wr * 2,     lane);
  v16bf q1 = lds_frag(sQ, wr * 2 + 1, lane);

  v8f acc[8];
#pragma unroll
  for (int i = 0; i < 8; ++i) { v8f z = {}; acc[i] = z; }

  for (int t0 = 0; t0 < SEQ; t0 += 64) {
    __syncthreads();
    // K tile -> B-frags: 256 x 32B copies
    {
      int key = tid >> 2, c = (tid >> 1) & 1, h2 = tid & 1;
      int dbase = c * 32 + h2 * 16;
      const uint4* g = (const uint4*)(K16 + (size_t)(b * SEQ + t0 + key) * DIM + hq * HDM + dbase);
      uint4* d = (uint4*)(sK + ((key >> 4) * 2 + c) * 512 + ((key & 15) + 16 * h2) * 16);
      d[0] = g[0]; d[1] = g[1];
    }
    // V tile (64 keys x 256 cols) -> B-frags: 1024 x 32B copies, async if available
#pragma unroll
    for (int e = tid; e < 1024; e += 256) {
      int col = e >> 2, c = (e >> 1) & 1, h2 = e & 1;
      const unsigned short* g =
          Vt + (size_t)(colbase + col) * MROWS + b * SEQ + t0 + c * 32 + h2 * 16;
      unsigned short* d = sV + (c * 16 + (col >> 4)) * 512 + ((col & 15) + 16 * h2) * 16;
#if HAVE_ASYNC
      __builtin_amdgcn_global_load_async_to_lds_b128(AS1V4(g), AS3V4(d), 0, 0);
      __builtin_amdgcn_global_load_async_to_lds_b128(AS1V4(g + 8), AS3V4(d + 8), 0, 0);
#else
      ((uint4*)d)[0] = ((const uint4*)g)[0];
      ((uint4*)d)[1] = ((const uint4*)g)[1];
#endif
    }
    async_wait_all();
    __syncthreads();
#pragma unroll
    for (int j = 0; j < 2; ++j) {                // score n-tiles owned by this wave
      int nt = wc * 2 + j;
      v8f z = {};
      v8f sc = wmma_bf16(q0, lds_frag(sK, nt * 2,     lane), z);
      sc     = wmma_bf16(q1, lds_frag(sK, nt * 2 + 1, lane), sc);
      int key = nt * 16 + nn;
      int c2 = key >> 5, kk = key & 31;
#pragma unroll
      for (int r = 0; r < 8; ++r) {              // P (normalized) -> A-frag layout in LDS
        float p = __expf(sc[r] - mi[r]) * li[r];
        sP[(wr * 2 + c2) * 512 + a_lane(r + 8 * hL, kk) * 16 + a_slot(kk)] = f2bf(p);
      }
    }
    __syncthreads();
    v16bf p0 = lds_frag(sP, wr * 2,     lane);
    v16bf p1 = lds_frag(sP, wr * 2 + 1, lane);
#pragma unroll
    for (int nt2 = 0; nt2 < 8; ++nt2) {
      int vf = wc * 8 + nt2;
      acc[nt2] = wmma_bf16(p0, lds_frag(sV, vf,      lane), acc[nt2]);
      acc[nt2] = wmma_bf16(p1, lds_frag(sV, 16 + vf, lane), acc[nt2]);
    }
  }
#pragma unroll
  for (int nt2 = 0; nt2 < 8; ++nt2) {
    int col = colbase + wc * 128 + nt2 * 16 + nn;
#pragma unroll
    for (int r = 0; r < 8; ++r) {
      int row = row0 + wr * 16 + r + 8 * hL;
      out16[((size_t)(b * SEQ + row) * NH + hq) * DIM + col] = f2bf(acc[nt2][r]);
    }
  }
}

// ------------------------------------------------- stage 2: rule composition
__global__ __launch_bounds__(128)
void k_compose(const unsigned short* __restrict__ out16, const unsigned short* __restrict__ Qv16,
               const float* __restrict__ Wkv, const float* __restrict__ bkv,
               unsigned short* __restrict__ o16) {
  int wglobal = blockIdx.x * 4 + (threadIdx.x >> 5);
  int lane = threadIdx.x & 31;
  int h = wglobal % NH;
  int sb = wglobal / NH;                         // b*SEQ + s
  size_t base = ((size_t)sb * NH + h) * DIM;
  float qv = bf2f(Qv16[(size_t)sb * (QKD * NH) + h * QKD + lane]);
  float tr[8];
#pragma unroll
  for (int r = 0; r < 8; ++r) {
    float a = bkv[lane];
    for (int d = 0; d < HDM; ++d)
      a += bf2f(out16[base + r * HDM + d]) * Wkv[d * QKD + lane];
    float p = qv * a;
    p += __shfl_xor(p, 1, 32);
    p += __shfl_xor(p, 2, 32);
    p += __shfl_xor(p, 4, 32);
    p += __shfl_xor(p, 8, 32);
    p += __shfl_xor(p, 16, 32);
    tr[r] = p;
  }
  float mx = tr[0];
#pragma unroll
  for (int r = 1; r < 8; ++r) mx = fmaxf(mx, tr[r]);
  float ex[8]; float s = 0.f;
#pragma unroll
  for (int r = 0; r < 8; ++r) { ex[r] = __expf(tr[r] - mx); s += ex[r]; }
  float inv = 1.f / s;
#pragma unroll
  for (int t = 0; t < 2; ++t) {
    int d = lane + 32 * t;
    float a = 0.f;
#pragma unroll
    for (int r = 0; r < 8; ++r) a += ex[r] * inv * bf2f(out16[base + r * HDM + d]);
    o16[(size_t)sb * DIM + h * HDM + d] = f2bf(a);
  }
}

// ------------------------------------------------- residual + LayerNorm
__global__ __launch_bounds__(256)
void k_ln(const float* __restrict__ resA, const float* __restrict__ addB,
          const float* __restrict__ g, const float* __restrict__ be,
          float* __restrict__ outF, unsigned short* __restrict__ outB, int mode) {
  __shared__ float rs[256], rq[256];
  int row = blockIdx.x, b = row / SEQ, s = row % SEQ;
  int tid = threadIdx.x;
  int c0 = tid * 2;
  size_t rbase = (size_t)row * DIM;
  size_t xbase = ((size_t)s * BSZ + b) * DIM;
  float v0, v1;
  if (mode == 0) {
    v0 = resA[xbase + c0]     + addB[rbase + c0];
    v1 = resA[xbase + c0 + 1] + addB[rbase + c0 + 1];
  } else {
    v0 = resA[rbase + c0]     + addB[rbase + c0];
    v1 = resA[rbase + c0 + 1] + addB[rbase + c0 + 1];
  }
  rs[tid] = v0 + v1;
  rq[tid] = v0 * v0 + v1 * v1;
  __syncthreads();
  for (int o = 128; o > 0; o >>= 1) {
    if (tid < o) { rs[tid] += rs[tid + o]; rq[tid] += rq[tid + o]; }
    __syncthreads();
  }
  float mu  = rs[0] * (1.f / DIM);
  float var = rq[0] * (1.f / DIM) - mu * mu;
  float rinv = rsqrtf(var + 1e-5f);
#pragma unroll
  for (int t = 0; t < 2; ++t) {
    int c = c0 + t;
    float v = t ? v1 : v0;
    float y = (v - mu) * rinv * g[c] + be[c];
    if (mode == 0) { outF[rbase + c] = y; outB[rbase + c] = f2bf(y); }
    else           outF[xbase + c] = y;
  }
}

// ---------------------------------------------------------------- launcher
extern "C" void kernel_launch(void* const* d_in, const int* in_sizes, int n_in,
                              void* d_out, int out_size, void* d_ws, size_t ws_size,
                              hipStream_t stream) {
  (void)in_sizes; (void)n_in; (void)out_size; (void)ws_size;
  const float* x   = (const float*)d_in[0];
  const float* Wq  = (const float*)d_in[1];
  const float* bq  = (const float*)d_in[2];
  const float* Wk  = (const float*)d_in[3];
  const float* bk  = (const float*)d_in[4];
  const float* Wv  = (const float*)d_in[5];
  const float* bv  = (const float*)d_in[6];
  const float* Wqv = (const float*)d_in[7];
  const float* bqv = (const float*)d_in[8];
  const float* Wkv = (const float*)d_in[9];
  const float* bkv = (const float*)d_in[10];
  const float* Wf  = (const float*)d_in[11];
  const float* bf_ = (const float*)d_in[12];
  const float* W1  = (const float*)d_in[13];
  const float* b1  = (const float*)d_in[14];
  const float* W2  = (const float*)d_in[15];
  const float* b2  = (const float*)d_in[16];
  const float* g1  = (const float*)d_in[17];
  const float* be1 = (const float*)d_in[18];
  const float* g2  = (const float*)d_in[19];
  const float* be2 = (const float*)d_in[20];

  char* ws = (char*)d_ws;
  size_t cur = 0;
  auto alloc = [&](size_t bytes) -> void* {
    void* p = ws + cur;
    cur = (cur + bytes + 255) & ~(size_t)255;
    return p;
  };
  unsigned short* WqT  = (unsigned short*)alloc((size_t)DIM * DIM * 2);
  unsigned short* WkT  = (unsigned short*)alloc((size_t)DIM * DIM * 2);
  unsigned short* WvT  = (unsigned short*)alloc((size_t)DIM * DIM * 2);
  unsigned short* WqvT = (unsigned short*)alloc((size_t)DIM * 256 * 2);
  unsigned short* WfT  = (unsigned short*)alloc((size_t)DIM * DIM * 2);
  unsigned short* W1T  = (unsigned short*)alloc((size_t)DIM * FFND * 2);
  unsigned short* W2T  = (unsigned short*)alloc((size_t)FFND * DIM * 2);
  unsigned short* xb16 = (unsigned short*)alloc((size_t)MROWS * DIM * 2);
  unsigned short* Q16  = (unsigned short*)alloc((size_t)MROWS * DIM * 2);
  unsigned short* K16  = (unsigned short*)alloc((size_t)MROWS * DIM * 2);
  unsigned short* Vt16 = (unsigned short*)alloc((size_t)MROWS * DIM * 2);
  unsigned short* Qv16 = (unsigned short*)alloc((size_t)MROWS * 256 * 2);
  float* mbuf          = (float*)alloc((size_t)BSZ * NH * SEQ * 4);
  float* lbuf          = (float*)alloc((size_t)BSZ * NH * SEQ * 4);
  unsigned short* out16= (unsigned short*)alloc((size_t)MROWS * NH * DIM * 2);
  unsigned short* o16  = (unsigned short*)alloc((size_t)MROWS * DIM * 2);
  float* y1            = (float*)alloc((size_t)MROWS * DIM * 4);
  float* h1f           = (float*)alloc((size_t)MROWS * DIM * 4);
  unsigned short* h1b  = (unsigned short*)alloc((size_t)MROWS * DIM * 2);
  unsigned short* f1b  = (unsigned short*)alloc((size_t)MROWS * FFND * 2);
  float* f2            = (float*)alloc((size_t)MROWS * DIM * 4);

  // pack (weights transposed to [N,K] for vectorized B staging)
  k_pack_x<<<(MROWS * DIM) / 256, 256, 0, stream>>>(x, xb16);
  k_pack_wT<<<(DIM * DIM + 255) / 256, 256, 0, stream>>>(Wq, WqT, DIM, DIM);
  k_pack_wT<<<(DIM * DIM + 255) / 256, 256, 0, stream>>>(Wk, WkT, DIM, DIM);
  k_pack_wT<<<(DIM * DIM + 255) / 256, 256, 0, stream>>>(Wv, WvT, DIM, DIM);
  k_pack_wT<<<(DIM * 256 + 255) / 256, 256, 0, stream>>>(Wqv, WqvT, DIM, 256);
  k_pack_wT<<<(DIM * DIM + 255) / 256, 256, 0, stream>>>(Wf, WfT, DIM, DIM);
  k_pack_wT<<<(DIM * FFND + 255) / 256, 256, 0, stream>>>(W1, W1T, DIM, FFND);
  k_pack_wT<<<(FFND * DIM + 255) / 256, 256, 0, stream>>>(W2, W2T, FFND, DIM);

  // projections
  const float qs  = 0.125f;               // 1/sqrt(64)
  const float qvs = 0.17677669529663689f; // 1/sqrt(32)
  k_gemm<<<dim3(DIM / 64, MROWS / 64), 128, 0, stream>>>(xb16, WqT,  bq,  qs,  DIM, DIM, 0, Q16,  nullptr);
  k_gemm<<<dim3(DIM / 64, MROWS / 64), 128, 0, stream>>>(xb16, WkT,  bk,  1.f, DIM, DIM, 0, K16,  nullptr);
  k_gemm<<<dim3(DIM / 64, MROWS / 64), 128, 0, stream>>>(xb16, WvT,  bv,  1.f, DIM, DIM, 3, Vt16, nullptr);
  k_gemm<<<dim3(256 / 64, MROWS / 64), 128, 0, stream>>>(xb16, WqvT, bqv, qvs, DIM, 256, 0, Qv16, nullptr);

  // attention
  k_attn_stats<<<dim3(SEQ / 64, BSZ * NH), 128, 0, stream>>>(Q16, K16, mbuf, lbuf);
  k_attn_out<<<dim3(SEQ / 64, BSZ * NH, 2), 256, 0, stream>>>(Q16, K16, Vt16, mbuf, lbuf, out16);

  // stage-2 composition
  k_compose<<<(MROWS * NH) / 4, 128, 0, stream>>>(out16, Qv16, Wkv, bkv, o16);

  // output projection + post-norm residual + FFN + final norm
  k_gemm<<<dim3(DIM / 64, MROWS / 64), 128, 0, stream>>>(o16, WfT, bf_, 1.f, DIM, DIM, 2, nullptr, y1);
  k_ln<<<MROWS, 256, 0, stream>>>(x, y1, g1, be1, h1f, h1b, 0);
  k_gemm<<<dim3(FFND / 64, MROWS / 64), 128, 0, stream>>>(h1b, W1T, b1, 1.f, DIM, FFND, 1, f1b, nullptr);
  k_gemm<<<dim3(DIM / 64, MROWS / 64), 128, 0, stream>>>(f1b, W2T, b2, 1.f, FFND, DIM, 2, nullptr, f2);
  k_ln<<<MROWS, 256, 0, stream>>>(h1f, f2, g2, be2, (float*)d_out, nullptr, 1);
}
